// MDHP_LSTM_67413806678224
// MI455X (gfx1250) — compile-verified, matrix operand order checked by reading, not run on hardware
//
#include <hip/hip_runtime.h>
#include <hip/hip_bf16.h>
#include <math.h>

// ---------------------------------------------------------------------------
// MDHP-LSTM on MI455X (gfx1250, wave32) — persistent-RNN + TDM design.
//   S=512, B=128, D=256, H=512, G=4H=2048
//
// One persistent kernel (16 WGs x 256 threads) runs the whole 512-step scan:
//   * transposed bf16 weights staged in LDS once (padded rows => conflict-free
//     ds_load_b128 fragment reads)
//   * per-step x panel [128x256] bf16 DMA'd into LDS by the Tensor Data Mover
//     (tensor_load_to_lds), with TDM pad_interval/pad_amount reproducing the
//     +16B/row anti-bank-conflict padding in hardware
//   * c-state / mdhp / bias held in registers for the whole scan
//   * h exchanged through a double-buffered global bf16 array, synchronized
//     with a device-scope atomic grid barrier once per step
// ---------------------------------------------------------------------------

typedef __bf16 bf16_t;
typedef __attribute__((ext_vector_type(16))) __bf16 v16bf;
typedef __attribute__((ext_vector_type(8)))  __bf16 v8bf;
typedef __attribute__((ext_vector_type(8)))  float  v8f;
typedef __attribute__((ext_vector_type(4)))  unsigned int v4u;
typedef __attribute__((ext_vector_type(8)))  int v8i_;
typedef __attribute__((ext_vector_type(4)))  int v4i_;

#define S_LEN 512
#define B_DIM 128
#define D_DIM 256
#define H_DIM 512
#define G_DIM (4 * H_DIM)

#define NWG   16            // persistent workgroups; each owns 32 hidden cols
#define LDW   (D_DIM + 8)   // padded LDS row stride (elems): +16B => +4 banks
#define LDU   (H_DIM + 8)
#define LDSW_BYTES ((size_t)128 * LDW * sizeof(bf16_t))   //  67,584 B
#define LDSU_BYTES ((size_t)128 * LDU * sizeof(bf16_t))   // 133,120 B
#define LDSX_BYTES ((size_t)B_DIM * LDW * sizeof(bf16_t)) //  67,584 B
#define LDS_BYTES  (LDSW_BYTES + LDSU_BYTES + LDSX_BYTES) // 268,288 B < 320 KB

__device__ __forceinline__ float sigmoid_f(float x) {
    return 1.0f / (1.0f + __expf(-x));
}

// 16x32 bf16 WMMA operand fragment from a row-major [rows, ld] matrix.
// ISA 7.12.2 (16-bit operands): lane L (idx=L%16, h=L/16):
//   elems 0..7 = K (8h)..(8h+7), elems 8..15 = K (16+8h)..(16+8h+7)
// A operand: idx = row. B operand (stored transposed [N,K]): idx = column.
template <typename PTR>
__device__ __forceinline__ v16bf load_frag(PTR base, int ld, int row0, int k0,
                                           int lane) {
    const int idx = lane & 15;
    const int hh  = lane >> 4;
    auto p = base + (size_t)(row0 + idx) * ld + k0 + 8 * hh;
    v8bf lo = *(const v8bf*)(p);
    v8bf hi = *(const v8bf*)(p + 16);
    return __builtin_shufflevector(lo, hi, 0, 1, 2, 3, 4, 5, 6, 7,
                                           8, 9, 10, 11, 12, 13, 14, 15);
}

// ---------------------------------------------------------------------------
// Setup kernels
// ---------------------------------------------------------------------------

__global__ void prep_x_kernel(const float* __restrict__ x,
                              bf16_t* __restrict__ xb, int n) {
    int i = blockIdx.x * blockDim.x + threadIdx.x;
    if (i < n) xb[i] = (bf16_t)x[i];
}

// Wt[g, d] = W_gate[d, j]   (g = gate*H + j), row-major [4H, D]
__global__ void prep_wt_kernel(const float* __restrict__ Wi,
                               const float* __restrict__ Wf,
                               const float* __restrict__ Wc,
                               const float* __restrict__ Wo,
                               bf16_t* __restrict__ Wt) {
    int i = blockIdx.x * blockDim.x + threadIdx.x;
    if (i >= G_DIM * D_DIM) return;
    int d = i % D_DIM;
    int g = i / D_DIM;
    int gate = g / H_DIM, j = g % H_DIM;
    const float* Wp = (gate == 0) ? Wi : (gate == 1) ? Wf : (gate == 2) ? Wc : Wo;
    Wt[i] = (bf16_t)Wp[(size_t)d * H_DIM + j];
}

// Ut[g, k] = U_gate[k, j], row-major [4H, H]
__global__ void prep_ut_kernel(const float* __restrict__ Ui,
                               const float* __restrict__ Uf,
                               const float* __restrict__ Uc,
                               const float* __restrict__ Uo,
                               bf16_t* __restrict__ Ut) {
    int i = blockIdx.x * blockDim.x + threadIdx.x;
    if (i >= G_DIM * H_DIM) return;
    int k = i % H_DIM;
    int g = i / H_DIM;
    int gate = g / H_DIM, j = g % H_DIM;
    const float* Up = (gate == 0) ? Ui : (gate == 1) ? Uf : (gate == 2) ? Uc : Uo;
    Ut[i] = (bf16_t)Up[(size_t)k * H_DIM + j];
}

__global__ void prep_bias_kernel(const float* __restrict__ bi,
                                 const float* __restrict__ bf,
                                 const float* __restrict__ bc,
                                 const float* __restrict__ bo,
                                 float* __restrict__ bias) {
    int g = blockIdx.x * blockDim.x + threadIdx.x;
    if (g >= G_DIM) return;
    int gate = g / H_DIM, j = g % H_DIM;
    const float* bp = (gate == 0) ? bi : (gate == 1) ? bf : (gate == 2) ? bc : bo;
    bias[g] = bp[j];
}

// mdhp[b, j] = tanh(alpha@A - tspan[b]*(beta@B) + theta@C)   [128, 512]
__global__ void mdhp_kernel(const float* __restrict__ alpha,
                            const float* __restrict__ beta,
                            const float* __restrict__ theta,
                            const float* __restrict__ tspan,
                            const float* __restrict__ A_m,
                            const float* __restrict__ B_m,
                            const float* __restrict__ C_m,
                            float* __restrict__ mdhp) {
    int i = blockIdx.x * blockDim.x + threadIdx.x;
    if (i >= B_DIM * H_DIM) return;
    int j = i % H_DIM;
    int b = i / H_DIM;
    float sA = 0.0f, sB = 0.0f, sC = 0.0f;
    for (int k = 0; k < 256; ++k) {
        sA += alpha[b * 256 + k] * A_m[(size_t)k * H_DIM + j];
        sB += beta [b * 256 + k] * B_m[(size_t)k * H_DIM + j];
    }
    for (int k = 0; k < 16; ++k)
        sC += theta[b * 16 + k] * C_m[(size_t)k * H_DIM + j];
    mdhp[i] = tanhf(sA - tspan[b] * sB + sC);
}

__global__ void init_state_kernel(const float* __restrict__ h0,
                                  bf16_t* __restrict__ hb,
                                  unsigned* __restrict__ bar) {
    int i = blockIdx.x * blockDim.x + threadIdx.x;
    if (i < B_DIM * H_DIM) hb[i] = (bf16_t)h0[i];
    if (i < 2) bar[i] = 0u;            // {counter, generation}
}

// ---------------------------------------------------------------------------
// Device-scope grid barrier (16 co-resident WGs; sense via generation count)
// ---------------------------------------------------------------------------
__device__ __forceinline__ void grid_sync(unsigned* counter, unsigned* gen) {
    __syncthreads();
    if (threadIdx.x == 0) {
        unsigned g0 = __hip_atomic_load(gen, __ATOMIC_RELAXED,
                                        __HIP_MEMORY_SCOPE_AGENT);
        unsigned arrived = __hip_atomic_fetch_add(counter, 1u, __ATOMIC_ACQ_REL,
                                                  __HIP_MEMORY_SCOPE_AGENT);
        if (arrived == NWG - 1) {
            __hip_atomic_store(counter, 0u, __ATOMIC_RELAXED,
                               __HIP_MEMORY_SCOPE_AGENT);
            // RELEASE: counter reset is visible before spinners pass
            __hip_atomic_fetch_add(gen, 1u, __ATOMIC_RELEASE,
                                   __HIP_MEMORY_SCOPE_AGENT);
        } else {
            while (__hip_atomic_load(gen, __ATOMIC_ACQUIRE,
                                     __HIP_MEMORY_SCOPE_AGENT) == g0)
                __builtin_amdgcn_s_sleep(2);
        }
    }
    __syncthreads();
}

// Issue a TDM load of the [128 x 256] bf16 x-panel at `gaddr` into LDS offset
// `lds_off`, with hardware padding of 4 dwords after every 128 dwords (one
// 512B row) -> LDS row stride of LDW=264 elements. ISA 8.3/8.4 D# packing.
// This toolchain exposes the 6-arg builtin:
//   (uint32x4 g0, int32x8 g1, int32x4 g2, int32x4 g3, int32x8 gx, i32 cpol)
__device__ __forceinline__ void tdm_load_x_panel(unsigned lds_off,
                                                 unsigned long long gaddr) {
    v4u g0;
    g0[0] = 1u;                                    // count=1, user descriptor
    g0[1] = lds_off;                               // lds_addr [63:32]
    g0[2] = (unsigned)(gaddr & 0xFFFFFFFFull);     // global_addr [95:64]
    g0[3] = (unsigned)((gaddr >> 32) & 0x01FFFFFFull) | (2u << 30); // +type=2
    v8i_ g1;
    g1[0] = (int)((1u << 16)     // data_size = 1 -> 2 bytes
                | (1u << 20)     // pad_enable
                | (6u << 22)     // pad_interval code 6 -> every 128 dwords
                | (3u << 25));   // pad_amount  code 3 -> 4 dwords (16 B)
    g1[1] = (int)(256u << 16);   // tensor_dim0[15:0] = 256 (elems)
    g1[2] = (int)(128u << 16);   // tensor_dim1[15:0] = 128
    g1[3] = (int)(256u << 16);   // tile_dim0 = 256
    g1[4] = (int)(128u);         // tile_dim1 = 128 (tile_dim2 = 0)
    g1[5] = (int)(256u);         // tensor_dim0_stride = 256 (elems)
    g1[6] = 0;
    g1[7] = 0;
    v4i_ gz4;
    gz4[0] = 0; gz4[1] = 0; gz4[2] = 0; gz4[3] = 0; // 2D: groups 2/3 unused
    v8i_ gz8;
#pragma unroll
    for (int q = 0; q < 8; ++q) gz8[q] = 0;
    __builtin_amdgcn_tensor_load_to_lds(g0, g1, gz4, gz4, gz8, 0);
}

// ---------------------------------------------------------------------------
// Persistent LSTM scan kernel: 16 WGs x 256 threads (8 waves)
//   WG w owns hidden cols [w*32, w*32+32) across all 4 gates.
//   wave (wm = wave&3, wn = wave>>2): rows wm*32..+32 (2 tiles), cols wn*16.
//   8 accumulators per wave: acc[rm][gate] (2 row tiles x 4 gates).
// ---------------------------------------------------------------------------
__global__ void __launch_bounds__(256) lstm_persistent_kernel(
    const bf16_t* __restrict__ x_bf,   // [S, B, D] bf16
    const bf16_t* __restrict__ Wt,     // [4H, D] bf16 transposed weights
    const bf16_t* __restrict__ Ut,     // [4H, H] bf16
    const float*  __restrict__ bias,   // [4H]
    const float*  __restrict__ mdhp,   // [B, H]
    const float*  __restrict__ c0,     // [B, H]
    bf16_t* __restrict__ hb0,          // [B, H] bf16 double buffer (t even in)
    bf16_t* __restrict__ hb1,          // [B, H] bf16 double buffer (t odd in)
    float*  __restrict__ out,          // [S, B, H]
    float*  __restrict__ hT,           // [B, H]
    float*  __restrict__ cT,           // [B, H]
    unsigned* __restrict__ bar) {      // {counter, gen}
    extern __shared__ __align__(16) char smem[];
    bf16_t* lds_w = (bf16_t*)smem;                            // [128][LDW]
    bf16_t* lds_u = (bf16_t*)(smem + LDSW_BYTES);             // [128][LDU]
    bf16_t* lds_x = (bf16_t*)(smem + LDSW_BYTES + LDSU_BYTES);// [128][LDW]
    // LDS byte offset of lds_x: generic LDS-aperture addresses carry the LDS
    // offset in their low 32 bits (ISA 10.2).
    const unsigned lds_x_off = (unsigned)(size_t)lds_x;

    const int wg   = blockIdx.x;       // 0..15
    const int lane = threadIdx.x & 31;
    const int wave = threadIdx.x >> 5;
    const int wm   = wave & 3;         // row group (32 rows)
    const int wn   = wave >> 2;        // col tile (16 cols of the WG's 32)

    // ---- Stage this WG's weight slices into LDS (once per scan) ----
    // Local row lr = gate*32 + jloc  <->  global row gate*512 + wg*32 + jloc
    for (int i = threadIdx.x; i < 128 * (D_DIM / 8); i += 256) {
        int lr = i / (D_DIM / 8);
        int c8 = (i % (D_DIM / 8)) * 8;
        int grow = (lr >> 5) * H_DIM + wg * 32 + (lr & 31);
        *(v8bf*)(lds_w + (size_t)lr * LDW + c8) =
            *(const v8bf*)(Wt + (size_t)grow * D_DIM + c8);
    }
    for (int i = threadIdx.x; i < 128 * (H_DIM / 8); i += 256) {
        int lr = i / (H_DIM / 8);
        int c8 = (i % (H_DIM / 8)) * 8;
        int grow = (lr >> 5) * H_DIM + wg * 32 + (lr & 31);
        *(v8bf*)(lds_u + (size_t)lr * LDU + c8) =
            *(const v8bf*)(Ut + (size_t)grow * H_DIM + c8);
    }

    // ---- Per-lane persistent state: column, bias, mdhp, c ----
    const int n   = lane & 15;
    const int hh  = lane >> 4;
    const int j   = wg * 32 + wn * 16 + n;      // hidden column in [0,512)
    float bg[4];
#pragma unroll
    for (int g = 0; g < 4; ++g) bg[g] = bias[g * H_DIM + j];

    float c_reg[2][8], m_reg[2][8];
#pragma unroll
    for (int rm = 0; rm < 2; ++rm)
#pragma unroll
        for (int e = 0; e < 8; ++e) {
            int row = wm * 32 + rm * 16 + e + 8 * hh;
            int idx = row * H_DIM + j;
            c_reg[rm][e] = c0[idx];
            m_reg[rm][e] = mdhp[idx];
        }

    __syncthreads();   // weights staged

    bf16_t* hb[2] = {hb0, hb1};

    // ---- Time scan ----
    for (int t = 0; t < S_LEN; ++t) {
        const bf16_t* hin  = hb[t & 1];
        bf16_t*       hout = hb[(t + 1) & 1];
        float*        ot_p = out + (size_t)t * B_DIM * H_DIM;

        // Wave 0 DMAs this step's x panel into LDS via the Tensor Data Mover;
        // TENSORcnt is per-wave, so wave 0 waits, then the WG barrier
        // publishes the panel to all waves.
        if (wave == 0) {
            tdm_load_x_panel(lds_x_off,
                             (unsigned long long)(size_t)(x_bf +
                                 (size_t)t * B_DIM * D_DIM));
            __builtin_amdgcn_s_wait_tensorcnt(0);
        }
        __syncthreads();

        v8f acc[2][4];
#pragma unroll
        for (int rm = 0; rm < 2; ++rm)
#pragma unroll
            for (int g = 0; g < 4; ++g)
#pragma unroll
                for (int e = 0; e < 8; ++e) acc[rm][g][e] = 0.0f;

        // Prefetch this step's h panel rows (global) while x GEMM runs.
        __builtin_prefetch(hin + (size_t)(wm * 32) * H_DIM, 0, 1);

        // K part 1: x_t @ W (K = 256); A and B both from LDS.
        for (int k0 = 0; k0 < D_DIM; k0 += 32) {
            v16bf a0 = load_frag((const bf16_t*)lds_x, LDW, wm * 32,      k0, lane);
            v16bf a1 = load_frag((const bf16_t*)lds_x, LDW, wm * 32 + 16, k0, lane);
#pragma unroll
            for (int g = 0; g < 4; ++g) {
                v16bf b = load_frag((const bf16_t*)lds_w, LDW,
                                    g * 32 + wn * 16, k0, lane);
                acc[0][g] = __builtin_amdgcn_wmma_f32_16x16x32_bf16(
                    false, a0, false, b, (short)0, acc[0][g], false, false);
                acc[1][g] = __builtin_amdgcn_wmma_f32_16x16x32_bf16(
                    false, a1, false, b, (short)0, acc[1][g], false, false);
            }
        }
        // K part 2: h @ U (K = 512); A from global, B from LDS.
        for (int k0 = 0; k0 < H_DIM; k0 += 32) {
            v16bf a0 = load_frag(hin, H_DIM, wm * 32,      k0, lane);
            v16bf a1 = load_frag(hin, H_DIM, wm * 32 + 16, k0, lane);
#pragma unroll
            for (int g = 0; g < 4; ++g) {
                v16bf b = load_frag((const bf16_t*)lds_u, LDU,
                                    g * 32 + wn * 16, k0, lane);
                acc[0][g] = __builtin_amdgcn_wmma_f32_16x16x32_bf16(
                    false, a0, false, b, (short)0, acc[0][g], false, false);
                acc[1][g] = __builtin_amdgcn_wmma_f32_16x16x32_bf16(
                    false, a1, false, b, (short)0, acc[1][g], false, false);
            }
        }

        // Fused gate epilogue on the accumulators; c stays in registers.
#pragma unroll
        for (int rm = 0; rm < 2; ++rm)
#pragma unroll
            for (int e = 0; e < 8; ++e) {
                int row = wm * 32 + rm * 16 + e + 8 * hh;
                int idx = row * H_DIM + j;
                float it = sigmoid_f(acc[rm][0][e] + bg[0]);
                float ft = sigmoid_f(acc[rm][1][e] + bg[1]);
                float ch = tanhf    (acc[rm][2][e] + bg[2]);
                float o_ = sigmoid_f(acc[rm][3][e] + bg[3]);
                float c = m_reg[rm][e] * (ft * c_reg[rm][e] + it * ch);
                float h = o_ * tanhf(c);
                c_reg[rm][e] = c;
                hout[idx] = (bf16_t)h;
                ot_p[idx] = h;
                if (t == S_LEN - 1) { hT[idx] = h; cT[idx] = c; }
            }

        // Make h visible device-wide, then barrier before next step reads it.
        __threadfence();
        grid_sync(bar, bar + 1);
        __threadfence();
    }
}

// ---------------------------------------------------------------------------
// Host launcher
// ---------------------------------------------------------------------------
extern "C" void kernel_launch(void* const* d_in, const int* in_sizes, int n_in,
                              void* d_out, int out_size, void* d_ws, size_t ws_size,
                              hipStream_t stream) {
    const float* x     = (const float*)d_in[0];
    const float* h0    = (const float*)d_in[1];
    const float* c0    = (const float*)d_in[2];
    const float* alpha = (const float*)d_in[3];
    const float* beta  = (const float*)d_in[4];
    const float* theta = (const float*)d_in[5];
    const float* tspan = (const float*)d_in[6];
    const float* A_m   = (const float*)d_in[7];
    const float* B_m   = (const float*)d_in[8];
    const float* C_m   = (const float*)d_in[9];
    const float* W_i = (const float*)d_in[10]; const float* U_i = (const float*)d_in[11]; const float* b_i = (const float*)d_in[12];
    const float* W_f = (const float*)d_in[13]; const float* U_f = (const float*)d_in[14]; const float* b_f = (const float*)d_in[15];
    const float* W_c = (const float*)d_in[16]; const float* U_c = (const float*)d_in[17]; const float* b_c = (const float*)d_in[18];
    const float* W_o = (const float*)d_in[19]; const float* U_o = (const float*)d_in[20]; const float* b_o = (const float*)d_in[21];

    float* out = (float*)d_out;                      // [S,B,H]
    float* hT  = out + (size_t)S_LEN * B_DIM * H_DIM;
    float* cT  = hT + (size_t)B_DIM * H_DIM;

    // Workspace carve-up (~36 MiB total).
    char* w = (char*)d_ws;
    bf16_t* x_bf = (bf16_t*)w; w += (size_t)S_LEN * B_DIM * D_DIM * sizeof(bf16_t);
    bf16_t* Wt   = (bf16_t*)w; w += (size_t)G_DIM * D_DIM * sizeof(bf16_t);
    bf16_t* Ut   = (bf16_t*)w; w += (size_t)G_DIM * H_DIM * sizeof(bf16_t);
    float*  bias = (float*)w;  w += (size_t)G_DIM * sizeof(float);
    float*  mdhp = (float*)w;  w += (size_t)B_DIM * H_DIM * sizeof(float);
    bf16_t* hb0  = (bf16_t*)w; w += (size_t)B_DIM * H_DIM * sizeof(bf16_t);
    bf16_t* hb1  = (bf16_t*)w; w += (size_t)B_DIM * H_DIM * sizeof(bf16_t);
    unsigned* bar = (unsigned*)w; w += 256;          // {counter, gen}

    const int TB = 256;
    {
        int nel = S_LEN * B_DIM * D_DIM;
        prep_x_kernel<<<(nel + TB - 1) / TB, TB, 0, stream>>>(x, x_bf, nel);
    }
    prep_wt_kernel<<<(G_DIM * D_DIM + TB - 1) / TB, TB, 0, stream>>>(W_i, W_f, W_c, W_o, Wt);
    prep_ut_kernel<<<(G_DIM * H_DIM + TB - 1) / TB, TB, 0, stream>>>(U_i, U_f, U_c, U_o, Ut);
    prep_bias_kernel<<<(G_DIM + TB - 1) / TB, TB, 0, stream>>>(b_i, b_f, b_c, b_o, bias);
    mdhp_kernel<<<(B_DIM * H_DIM + TB - 1) / TB, TB, 0, stream>>>(
        alpha, beta, theta, tspan, A_m, B_m, C_m, mdhp);
    init_state_kernel<<<(B_DIM * H_DIM + TB - 1) / TB, TB, 0, stream>>>(h0, hb0, bar);

    // Single persistent kernel runs the whole 512-step scan.
    lstm_persistent_kernel<<<NWG, 256, LDS_BYTES, stream>>>(
        x_bf, Wt, Ut, bias, mdhp, c0, hb0, hb1, out, hT, cT, bar);
}
